// DecoderBlock_31147102831149
// MI455X (gfx1250) — compile-verified, matrix operand order checked by reading, not run on hardware
//
#include <hip/hip_runtime.h>
#include <math.h>

// ---------------- problem constants ----------------
#define BB 4
#define SS 2048
#define EE 1024
#define HH 16
#define DD 64
#define FF 4096
#define MM (BB * SS)          // 8192 token rows

typedef unsigned short u16;
typedef __attribute__((ext_vector_type(16))) __bf16 v16bf;
typedef __attribute__((ext_vector_type(8)))  float  v8f;

union Frag16 {               // one WMMA bf16 operand: 16 bf16 = 32B = 8 VGPRs
    uint4 q[2];
    u16   h[16];
    v16bf v;
};

__device__ __forceinline__ u16 f2bf(float f) {
    union { float f; unsigned u; } x; x.f = f;
    unsigned r = x.u + 0x7FFFu + ((x.u >> 16) & 1u);   // round-to-nearest-even
    return (u16)(r >> 16);
}

__device__ __forceinline__ v8f wmma_bf16(const Frag16& a, const Frag16& b, v8f c) {
    return __builtin_amdgcn_wmma_f32_16x16x32_bf16(
        /*neg_a=*/false, a.v, /*neg_b=*/false, b.v,
        /*c_mod=*/(short)0, c, /*reuse_a=*/false, /*reuse_b=*/false);
}

// per-lane async copy: 16B global -> 16B LDS, tracked by ASYNCcnt
__device__ __forceinline__ void async_cp16(const u16* gptr, u16* lptr) {
    unsigned lds = (unsigned)(size_t)lptr;     // flat shared ptr low 32 bits == LDS addr
    asm volatile("global_load_async_to_lds_b128 %0, %1, off"
                 :: "v"(lds), "v"(gptr) : "memory");
}
__device__ __forceinline__ void wait_async0() {
    asm volatile("s_wait_asynccnt 0x0" ::: "memory");
}

// ---------------- fp32 -> bf16 weight convert ----------------
__global__ __launch_bounds__(256) void conv_bf16_kernel(const float* __restrict__ src,
                                                        u16* __restrict__ dst, int n) {
    int i = blockIdx.x * 256 + threadIdx.x;
    if (i < n) dst[i] = f2bf(src[i]);
}

// ---------------- bias concat for fused QKV ----------------
__global__ __launch_bounds__(256) void bias_cat_kernel(const float* __restrict__ bq,
                                                       const float* __restrict__ bk,
                                                       const float* __restrict__ bv,
                                                       float* __restrict__ dst) {
    int i = blockIdx.x * 256 + threadIdx.x;
    if (i < 3 * EE) {
        float v = (i < EE) ? bq[i] : (i < 2 * EE) ? bk[i - EE] : bv[i - 2 * EE];
        dst[i] = v;
    }
}

// ---------------- layernorm (fp32 in -> bf16 out) ----------------
__global__ __launch_bounds__(256) void ln_kernel(const float* __restrict__ x,
                                                 const float* __restrict__ g,
                                                 const float* __restrict__ b,
                                                 u16* __restrict__ out) {
    __shared__ float red[256];
    const int row = blockIdx.x;
    const int tid = threadIdx.x;
    const float* xr = x + (size_t)row * EE;

    float s = 0.f, ss = 0.f, v[4];
#pragma unroll
    for (int j = 0; j < 4; ++j) {
        v[j] = xr[tid + 256 * j];
        s += v[j]; ss += v[j] * v[j];
    }
    red[tid] = s; __syncthreads();
    for (int o = 128; o > 0; o >>= 1) { if (tid < o) red[tid] += red[tid + o]; __syncthreads(); }
    float mean = red[0] * (1.0f / EE);
    __syncthreads();
    red[tid] = ss; __syncthreads();
    for (int o = 128; o > 0; o >>= 1) { if (tid < o) red[tid] += red[tid + o]; __syncthreads(); }
    float var = fmaxf(red[0] * (1.0f / EE) - mean * mean, 0.f);
    float rstd = rsqrtf(var + 1e-5f);

    u16* orow = out + (size_t)row * EE;
#pragma unroll
    for (int j = 0; j < 4; ++j) {
        int e = tid + 256 * j;
        orow[e] = f2bf((v[j] - mean) * rstd * g[e] + b[e]);
    }
}

// ---------------- tiled WMMA GEMM:  Y = epi(X[M,K] * W[N,K]^T + bias[N]) ----------------
// Block tile 128x128, 8 waves (4x2), wave tile 32x64 (2x4 WMMA accums), K-step 32.
// Double-buffered LDS fed by GLOBAL_LOAD_ASYNC_TO_LDS_B128 (ASYNCcnt pipeline).
// EPI: 0 = store bf16, 1 = exact GELU -> bf16, 2 = +resid(fp32) -> fp32.
#define LDSTR 40   // bf16 elements per LDS row (80B: 16B-aligned, conflict-free step)

template <int EPI>
__global__ __launch_bounds__(256) void gemm_kernel(const u16*  __restrict__ A,
                                                   const u16*  __restrict__ W,
                                                   const float* __restrict__ bias,
                                                   const float* __restrict__ resid,
                                                   u16*   __restrict__ outB,
                                                   float* __restrict__ outF,
                                                   int N, int K) {
    __shared__ u16 As[2][128 * LDSTR];
    __shared__ u16 Bs[2][128 * LDSTR];

    const int tid = threadIdx.x;
    const int bm = blockIdx.y * 128, bn = blockIdx.x * 128;
    const int wid  = tid >> 5;
    const int lane = tid & 31;
    const int l16  = lane & 15;
    const int hf   = (lane >> 4) & 1;
    const int wm = (wid & 3) * 32;
    const int wn = (wid >> 2) * 64;

    // this thread's two 16B chunks of the 128x32 tile: chunk -> (row, col-chunk)
    const int r0 = (tid + 0)   >> 2, c0 = (tid + 0)   & 3;
    const int r1 = (tid + 256) >> 2, c1 = (tid + 256) & 3;

    v8f acc[2][4];
#pragma unroll
    for (int mt = 0; mt < 2; ++mt)
#pragma unroll
        for (int nt = 0; nt < 4; ++nt)
#pragma unroll
            for (int i = 0; i < 8; ++i) acc[mt][nt][i] = 0.f;

    const int nk = K / 32;
    // prologue: async-stage tile 0 into buffer 0
    {
        async_cp16(&A[(size_t)(bm + r0) * K + c0 * 8], &As[0][r0 * LDSTR + c0 * 8]);
        async_cp16(&W[(size_t)(bn + r0) * K + c0 * 8], &Bs[0][r0 * LDSTR + c0 * 8]);
        async_cp16(&A[(size_t)(bm + r1) * K + c1 * 8], &As[0][r1 * LDSTR + c1 * 8]);
        async_cp16(&W[(size_t)(bn + r1) * K + c1 * 8], &Bs[0][r1 * LDSTR + c1 * 8]);
    }

    for (int t = 0; t < nk; ++t) {
        wait_async0();          // my wave's tile-t copies landed in LDS
        __syncthreads();        // everyone's tile-t copies visible

        if (t + 1 < nk) {       // async-stage tile t+1 into the other buffer
            const int nb = (t + 1) & 1;
            const int k0 = (t + 1) * 32;
            async_cp16(&A[(size_t)(bm + r0) * K + k0 + c0 * 8], &As[nb][r0 * LDSTR + c0 * 8]);
            async_cp16(&W[(size_t)(bn + r0) * K + k0 + c0 * 8], &Bs[nb][r0 * LDSTR + c0 * 8]);
            async_cp16(&A[(size_t)(bm + r1) * K + k0 + c1 * 8], &As[nb][r1 * LDSTR + c1 * 8]);
            async_cp16(&W[(size_t)(bn + r1) * K + k0 + c1 * 8], &Bs[nb][r1 * LDSTR + c1 * 8]);
        }

        const int cb = t & 1;
        Frag16 af[2], bf[4];
#pragma unroll
        for (int mt = 0; mt < 2; ++mt) {
            int r = wm + mt * 16 + l16;
            af[mt].q[0] = *(const uint4*)&As[cb][r * LDSTR + hf * 8];
            af[mt].q[1] = *(const uint4*)&As[cb][r * LDSTR + hf * 8 + 16];
        }
#pragma unroll
        for (int nt = 0; nt < 4; ++nt) {
            int r = wn + nt * 16 + l16;
            bf[nt].q[0] = *(const uint4*)&Bs[cb][r * LDSTR + hf * 8];
            bf[nt].q[1] = *(const uint4*)&Bs[cb][r * LDSTR + hf * 8 + 16];
        }
#pragma unroll
        for (int mt = 0; mt < 2; ++mt)
#pragma unroll
            for (int nt = 0; nt < 4; ++nt)
                acc[mt][nt] = wmma_bf16(af[mt], bf[nt], acc[mt][nt]);
        // no trailing barrier: tile t+2's async writes are gated by the next
        // top-of-loop barrier, and our ds reads are consumed before the WMMAs.
    }

    // epilogue: C layout -> row = i + 8*hf, col = l16
#pragma unroll
    for (int mt = 0; mt < 2; ++mt)
#pragma unroll
        for (int nt = 0; nt < 4; ++nt)
#pragma unroll
            for (int i = 0; i < 8; ++i) {
                int gr = bm + wm + mt * 16 + i + 8 * hf;
                int gc = bn + wn + nt * 16 + l16;
                float v = acc[mt][nt][i] + bias[gc];
                if (EPI == 0) {
                    outB[(size_t)gr * N + gc] = f2bf(v);
                } else if (EPI == 1) {
                    float ge = 0.5f * v * (1.0f + erff(v * 0.70710678118654752f));
                    outB[(size_t)gr * N + gc] = f2bf(ge);
                } else {
                    outF[(size_t)gr * N + gc] = v + resid[(size_t)gr * N + gc];
                }
            }
}

// ---------------- flash attention (causal), bf16 WMMA, fp32 online softmax ----------------
// qkv: [M, 3E] bf16, cols 0..1023=q, 1024..2047=k, 2048..3071=v (== [b,s,h,d]).
// ctx: [M, E] bf16. Block = 128 thr (4 waves); each wave owns 16 query rows.
// V tiles are async-staged to wave-private LDS (overlapped with the QK^T WMMAs).
#define VSTR 72    // V stage row stride in bf16 (144B, 16B-aligned)

__global__ __launch_bounds__(128) void attn_kernel(const u16* __restrict__ qkv,
                                                   u16* __restrict__ ctx) {
    __shared__ u16 Plds[4][16 * 32];    // per-wave probability tile (bf16)
    __shared__ u16 Vlds[4][32 * VSTR];  // per-wave V tile stage (32 rows x 64 cols)

    const int b = blockIdx.z, h = blockIdx.y;
    const int w = threadIdx.x >> 5;
    const int lane = threadIdx.x & 31;
    const int l16 = lane & 15;
    const int hf  = (lane >> 4) & 1;
    const int qb = (blockIdx.x * 4 + w) * 16;
    const size_t rowstr = 3 * EE;

    // preload Q A-fragments (two K-steps over D=64)
    Frag16 qf[2];
#pragma unroll
    for (int ks = 0; ks < 2; ++ks) {
        size_t base = (size_t)(b * SS + qb + l16) * rowstr + h * DD + 32 * ks + hf * 8;
        qf[ks].q[0] = *(const uint4*)&qkv[base];
        qf[ks].q[1] = *(const uint4*)&qkv[base + 16];
    }

    float m_arr[8], l_arr[8];
    v8f cacc[4];
#pragma unroll
    for (int i = 0; i < 8; ++i) { m_arr[i] = -3.0e38f; l_arr[i] = 0.f; }
#pragma unroll
    for (int n = 0; n < 4; ++n)
#pragma unroll
        for (int i = 0; i < 8; ++i) cacc[n][i] = 0.f;

    u16* Vw = Vlds[w];
    u16* P  = Plds[w];

    for (int kb = 0; kb <= qb + 15; kb += 32) {
        // ---- kick off async V-tile copy (32 rows x 64 bf16), overlaps QK^T ----
        // prior iteration's V ds-reads were consumed by its WMMAs before this issues.
#pragma unroll
        for (int j = 0; j < 8; ++j) {
            int chunk = lane + 32 * j;              // 0..255
            int vr = chunk >> 3, cc = chunk & 7;
            async_cp16(&qkv[(size_t)(b * SS + kb + vr) * rowstr + 2 * EE + h * DD + cc * 8],
                       &Vw[vr * VSTR + cc * 8]);
        }

        // ---- scores = Q * K^T (two 16-col subtiles) ----
        v8f sc[2];
#pragma unroll
        for (int nt = 0; nt < 2; ++nt) {
#pragma unroll
            for (int i = 0; i < 8; ++i) sc[nt][i] = 0.f;
#pragma unroll
            for (int ks = 0; ks < 2; ++ks) {
                Frag16 kf;
                size_t base = (size_t)(b * SS + kb + nt * 16 + l16) * rowstr
                              + EE + h * DD + 32 * ks + hf * 8;
                kf.q[0] = *(const uint4*)&qkv[base];
                kf.q[1] = *(const uint4*)&qkv[base + 16];
                sc[nt] = wmma_bf16(qf[ks], kf, sc[nt]);
            }
        }
        // ---- scale, causal mask, online softmax stats ----
        float mnew[8];
#pragma unroll
        for (int i = 0; i < 8; ++i) {
            int r = qb + i + 8 * hf;
            float s0 = sc[0][i] * 0.125f;
            float s1 = sc[1][i] * 0.125f;
            if (kb + l16 > r)       s0 = -3.0e38f;
            if (kb + 16 + l16 > r)  s1 = -3.0e38f;
            sc[0][i] = s0; sc[1][i] = s1;
            float mx = fmaxf(s0, s1);
#pragma unroll
            for (int o = 1; o < 16; o <<= 1) mx = fmaxf(mx, __shfl_xor(mx, o, 32));
            mnew[i] = fmaxf(m_arr[i], mx);
        }
#pragma unroll
        for (int i = 0; i < 8; ++i) {
            float corr = __expf(m_arr[i] - mnew[i]);
            float p0 = __expf(sc[0][i] - mnew[i]);
            float p1 = __expf(sc[1][i] - mnew[i]);
            sc[0][i] = p0; sc[1][i] = p1;
            float rs = p0 + p1;
#pragma unroll
            for (int o = 1; o < 16; o <<= 1) rs += __shfl_xor(rs, o, 32);
            l_arr[i] = l_arr[i] * corr + rs;
            m_arr[i] = mnew[i];
#pragma unroll
            for (int n = 0; n < 4; ++n) cacc[n][i] *= corr;
        }
        // ---- P (C-layout) -> LDS bf16 -> reload in A-fragment layout ----
#pragma unroll
        for (int i = 0; i < 8; ++i) {
            int r = i + 8 * hf;
            P[r * 32 + l16]      = f2bf(sc[0][i]);
            P[r * 32 + 16 + l16] = f2bf(sc[1][i]);
        }
        asm volatile("s_wait_dscnt 0x0" ::: "memory");
        Frag16 pf;
        {
            int off = l16 * 32 + hf * 8;
            pf.q[0] = *(const uint4*)&P[off];
            pf.q[1] = *(const uint4*)&P[off + 16];
        }
        // ---- ctx += P * V  (V gathered from LDS in B-fragment order) ----
        wait_async0();          // wave-private tile: no workgroup barrier needed
#pragma unroll
        for (int n = 0; n < 4; ++n) {
            Frag16 vf;
#pragma unroll
            for (int j = 0; j < 8; ++j) {
                vf.h[j]     = Vw[(8 * hf + j) * VSTR      + n * 16 + l16];
                vf.h[8 + j] = Vw[(16 + 8 * hf + j) * VSTR + n * 16 + l16];
            }
            cacc[n] = wmma_bf16(pf, vf, cacc[n]);
        }
    }

    // ---- normalize and store ctx (bf16, [M,E]) ----
#pragma unroll
    for (int i = 0; i < 8; ++i) {
        float inv = 1.0f / l_arr[i];
        int r = qb + i + 8 * hf;
#pragma unroll
        for (int n = 0; n < 4; ++n)
            ctx[(size_t)(b * SS + r) * EE + h * DD + n * 16 + l16] = f2bf(cacc[n][i] * inv);
    }
}

// ---------------- launch ----------------
extern "C" void kernel_launch(void* const* d_in, const int* in_sizes, int n_in,
                              void* d_out, int out_size, void* d_ws, size_t ws_size,
                              hipStream_t stream) {
    (void)in_sizes; (void)n_in; (void)out_size; (void)ws_size;
    const float* x    = (const float*)d_in[0];
    // d_in[1] = mask (causal; implemented analytically)
    const float* ln1g = (const float*)d_in[2];
    const float* ln1b = (const float*)d_in[3];
    const float* Wq   = (const float*)d_in[4];
    const float* bq   = (const float*)d_in[5];
    const float* Wk   = (const float*)d_in[6];
    const float* bk   = (const float*)d_in[7];
    const float* Wv   = (const float*)d_in[8];
    const float* bv   = (const float*)d_in[9];
    const float* Wo   = (const float*)d_in[10];
    const float* bo   = (const float*)d_in[11];
    const float* ln2g = (const float*)d_in[12];
    const float* ln2b = (const float*)d_in[13];
    const float* W1   = (const float*)d_in[14];
    const float* b1   = (const float*)d_in[15];
    const float* W2   = (const float*)d_in[16];
    const float* b2   = (const float*)d_in[17];
    float* out = (float*)d_out;

    // workspace layout (bytes)
    char* ws = (char*)d_ws;
    const size_t EE2 = (size_t)EE * EE;
    u16*   wqkv = (u16*)(ws);                                    // 3E*E bf16
    u16*   wo   = (u16*)(ws + 6291456);                          // E*E
    u16*   w1   = (u16*)(ws + 8388608);                          // F*E
    u16*   w2   = (u16*)(ws + 16777216);                         // E*F
    float* bqkv = (float*)(ws + 25165824);                       // 3E fp32
    u16*   h1   = (u16*)(ws + 25178112);                         // M*E bf16 (reused as ctx)
    u16*   qkv  = (u16*)(ws + 41955328);                         // M*3E bf16 (reused as h2)
    float* x2   = (float*)(ws + 92286976);                       // M*E fp32
    u16*   g    = (u16*)(ws + 125841408);                        // M*F bf16
    u16*   ctx  = h1;
    u16*   h2   = qkv;

    // 1) convert weights to bf16 (QKV packed into one [3E,E])
    conv_bf16_kernel<<<EE2 / 256, 256, 0, stream>>>(Wq, wqkv,           (int)EE2);
    conv_bf16_kernel<<<EE2 / 256, 256, 0, stream>>>(Wk, wqkv + EE2,     (int)EE2);
    conv_bf16_kernel<<<EE2 / 256, 256, 0, stream>>>(Wv, wqkv + 2 * EE2, (int)EE2);
    conv_bf16_kernel<<<EE2 / 256, 256, 0, stream>>>(Wo, wo,             (int)EE2);
    conv_bf16_kernel<<<(size_t)FF * EE / 256, 256, 0, stream>>>(W1, w1, FF * EE);
    conv_bf16_kernel<<<(size_t)EE * FF / 256, 256, 0, stream>>>(W2, w2, EE * FF);
    bias_cat_kernel<<<12, 256, 0, stream>>>(bq, bk, bv, bqkv);

    // 2) LN1 -> h1 (bf16)
    ln_kernel<<<MM, 256, 0, stream>>>(x, ln1g, ln1b, h1);

    // 3) fused QKV GEMM: qkv = h1 * Wqkv^T + bqkv   [M, 3E]
    gemm_kernel<0><<<dim3(3 * EE / 128, MM / 128), 256, 0, stream>>>(
        h1, wqkv, bqkv, nullptr, qkv, nullptr, 3 * EE, EE);

    // 4) causal flash attention -> ctx (bf16)
    attn_kernel<<<dim3(SS / 64, HH, BB), 128, 0, stream>>>(qkv, ctx);

    // 5) out projection + residual: x2 = ctx * Wo^T + bo + x   (fp32)
    gemm_kernel<2><<<dim3(EE / 128, MM / 128), 256, 0, stream>>>(
        ctx, wo, bo, x, nullptr, x2, EE, EE);

    // 6) LN2 -> h2 (bf16)
    ln_kernel<<<MM, 256, 0, stream>>>(x2, ln2g, ln2b, h2);

    // 7) FFN1 + exact GELU: g = gelu(h2 * W1^T + b1)   [M, F] bf16
    gemm_kernel<1><<<dim3(FF / 128, MM / 128), 256, 0, stream>>>(
        h2, w1, b1, nullptr, g, nullptr, FF, EE);

    // 8) FFN2 + residual: out = g * W2^T + b2 + x2   (fp32)
    gemm_kernel<2><<<dim3(EE / 128, MM / 128), 256, 0, stream>>>(
        g, w2, b2, x2, nullptr, out, EE, FF);
}
